// Physics_constraint_SSM_40578851012653
// MI455X (gfx1250) — compile-verified, matrix operand order checked by reading, not run on hardware
//
#include <hip/hip_runtime.h>
#include <hip/hip_bf16.h>
#include <math.h>

// ---------------------------------------------------------------------------
// Physics-constraint SSM (S5 x2 + head + batched 64x64 solve + z-scan)
// Target: MI455X / gfx1250 (wave32, WMMA, TDM async tensor loads).
// Workspace use (~300 MB): bufA(268MB) | xA | xB | fx | bu_re | bu_im |
//                          Lbar_re/im | Bbar_re/im | h1(bf16) | W2T(bf16)
// ---------------------------------------------------------------------------

#define BSZ   32
#define LSEQ  512
#define NDIM  64
#define PDIM  128
#define HDIM  128
#define NTOK  (BSZ*LSEQ)       // 16384
#define NN    (NDIM*NDIM)      // 4096

typedef __attribute__((ext_vector_type(16))) __bf16 v16bf;
typedef __attribute__((ext_vector_type(8)))  float  v8f;

union BfFrag { v16bf v; uint4 q[2]; };

#ifndef __has_builtin
#define __has_builtin(x) 0
#endif
#if __has_builtin(__builtin_amdgcn_tensor_load_to_lds) && \
    __has_builtin(__builtin_amdgcn_s_wait_tensorcnt)
#define USE_TDM 1
#else
#define USE_TDM 0
#endif

#if USE_TDM
typedef unsigned int v4u __attribute__((ext_vector_type(4)));
typedef int          v4i __attribute__((ext_vector_type(4)));
typedef int          v8i __attribute__((ext_vector_type(8)));

// One-shot TDM DMA: contiguous `nElem` fp32 from global -> LDS.
// D# per CDNA5 ISA ch.8: group0 {count=1, lds_addr, global_addr, type=2},
// group1 {data_size=4B, tensor_dim0=tile_dim0=nElem, dim1=1, stride0=nElem},
// groups 2/3 zero (<=2D tensor). Issued once per wave; EXEC ignored.
static __device__ __forceinline__ void tdm_load_f32(const float* gsrc,
                                                    void* lds_dst,
                                                    unsigned int nElem)
{
    unsigned long long ga = (unsigned long long)gsrc;          // 57-bit byte VA
    unsigned int laddr = (unsigned int)(unsigned long long)lds_dst; // LDS offset
    v4u g0;
    g0[0] = 1u;                                                // count=1
    g0[1] = laddr;                                             // lds_addr
    g0[2] = (unsigned int)(ga & 0xFFFFFFFFu);                  // global[31:0]
    g0[3] = (unsigned int)((ga >> 32) & 0x01FFFFFFu) | (2u << 30); // +type=2
    v8i g1;
    g1[0] = (int)(2u << 16);                 // wg_mask=0, data_size=2 (4B)
    g1[1] = (int)((nElem & 0xFFFFu) << 16);  // tensor_dim0[15:0] @ bits63:48
    g1[2] = (int)(((nElem >> 16) & 0xFFFFu)) | (int)(1u << 16); // dim0 hi | dim1=1
    g1[3] = (int)((nElem & 0xFFFFu) << 16);  // tile_dim0 @ bits127:112
    g1[4] = 1;                               // tile_dim1=1, tile_dim2=0
    g1[5] = (int)nElem;                      // tensor_dim0_stride[31:0]
    g1[6] = 0;
    g1[7] = 0;
    v4i z4 = {0, 0, 0, 0};
#if __clang_major__ >= 23
    v8i z8 = {0, 0, 0, 0, 0, 0, 0, 0};
    __builtin_amdgcn_tensor_load_to_lds(g0, g1, z4, z4, z8, 0);
#else
    __builtin_amdgcn_tensor_load_to_lds(g0, g1, z4, z4, 0);
#endif
}
#endif // USE_TDM

static __device__ __forceinline__ unsigned short f2bf(float f) {
    unsigned int u = __float_as_uint(f);
    unsigned int r = (u + 0x7FFFu + ((u >> 16) & 1u)) >> 16;   // RNE
    return (unsigned short)r;
}
static __device__ __forceinline__ float gelu_exact(float x) {
    return 0.5f * x * (1.0f + erff(x * 0.70710678118654752f));
}

// ---------------------------------------------------------------------------
// LayerNorm over last dim (N=64). One block (64 thr) per token row.
// ---------------------------------------------------------------------------
__global__ void k_layernorm(const float* __restrict__ xin,
                            const float* __restrict__ w,
                            const float* __restrict__ b,
                            float* __restrict__ fxout)
{
    int tok = blockIdx.x, t = threadIdx.x;
    __shared__ float s[NDIM];
    __shared__ float red[2];
    float v = xin[(size_t)tok * NDIM + t];
    s[t] = v;
    __syncthreads();
    if (t == 0) {
        float m = 0.f;
        for (int h = 0; h < NDIM; ++h) m += s[h];
        m *= (1.0f / NDIM);
        float var = 0.f;
        for (int h = 0; h < NDIM; ++h) { float d = s[h] - m; var += d * d; }
        var *= (1.0f / NDIM);
        red[0] = m; red[1] = rsqrtf(var + 1e-5f);
    }
    __syncthreads();
    fxout[(size_t)tok * NDIM + t] = (v - red[0]) * red[1] * w[t] + b[t];
}

// ---------------------------------------------------------------------------
// Discretize: Lbar = exp(Lam*step), Bbar = ((Lbar-1)/Lam) * Bc.  1 block x 128.
// ---------------------------------------------------------------------------
__global__ void k_prep(const float* __restrict__ Lam_re,
                       const float* __restrict__ Lam_im,
                       const float* __restrict__ log_step,
                       const float* __restrict__ Bre,
                       const float* __restrict__ Bim,
                       float* __restrict__ Lbar_re, float* __restrict__ Lbar_im,
                       float* __restrict__ Bbar_re, float* __restrict__ Bbar_im)
{
    int p = threadIdx.x;                       // 0..127
    float step = expf(log_step[p]);
    float lre  = -expf(Lam_re[p]);
    float lim  = Lam_im[p];
    float er   = expf(lre * step);
    float cbr  = er * cosf(lim * step);
    float cbi  = er * sinf(lim * step);
    Lbar_re[p] = cbr; Lbar_im[p] = cbi;
    float nr = cbr - 1.0f, ni = cbi;
    float den = lre * lre + lim * lim;
    float cr = (nr * lre + ni * lim) / den;    // (Lbar-1)/Lam, complex div
    float ci = (ni * lre - nr * lim) / den;
    for (int h = 0; h < NDIM; ++h) {
        float br = Bre[p * NDIM + h], bi = Bim[p * NDIM + h];
        Bbar_re[p * NDIM + h] = cr * br - ci * bi;
        Bbar_im[p * NDIM + h] = cr * bi + ci * br;
    }
}

// ---------------------------------------------------------------------------
// Bu[b,l,p] = sum_h Bbar[p,h] * fx[b,l,h]   (complex). One block (128) / token.
// ---------------------------------------------------------------------------
__global__ void k_bu(const float* __restrict__ fx,
                     const float* __restrict__ Bbr,
                     const float* __restrict__ Bbi,
                     float* __restrict__ bur, float* __restrict__ bui)
{
    int tok = blockIdx.x, p = threadIdx.x;
    __shared__ float sfx[NDIM];
    if (p < NDIM) sfx[p] = fx[(size_t)tok * NDIM + p];
    __syncthreads();
    const float* br = Bbr + p * NDIM;
    const float* bi = Bbi + p * NDIM;
    float ar = 0.f, ai = 0.f;
    #pragma unroll 8
    for (int h = 0; h < NDIM; ++h) { float u = sfx[h]; ar += br[h] * u; ai += bi[h] * u; }
    bur[(size_t)tok * PDIM + p] = ar;
    bui[(size_t)tok * PDIM + p] = ai;
}

// ---------------------------------------------------------------------------
// Complex first-order scan along L, in place over bu. Thread = (b,p).
// ---------------------------------------------------------------------------
__global__ void k_s5scan(const float* __restrict__ Lbr,
                         const float* __restrict__ Lbi,
                         float* __restrict__ bur, float* __restrict__ bui)
{
    int b = blockIdx.x, p = threadIdx.x;
    float lr = Lbr[p], li = Lbi[p];
    float xr = 0.f, xi = 0.f;
    size_t base = ((size_t)b * LSEQ) * PDIM + p;
    for (int l = 0; l < LSEQ; ++l) {
        size_t idx = base + (size_t)l * PDIM;
        float ur = bur[idx], ui = bui[idx];
        float nr = lr * xr - li * xi + ur;
        float ni = lr * xi + li * xr + ui;
        xr = nr; xi = ni;
        bur[idx] = xr; bui[idx] = xi;
    }
}

// ---------------------------------------------------------------------------
// y = 2*Re(Cc @ xs) + D*u ; x1 = gelu(y)+fx ; fx2 = LN2(x1) ;
// GLU FFN ; out = (v*gelu(g)) @ Wdec + fx2.   One block (128 thr) per token.
// ---------------------------------------------------------------------------
__global__ void k_post(const float* __restrict__ xsre, const float* __restrict__ xsim,
                       const float* __restrict__ fx,
                       const float* __restrict__ Cre,  const float* __restrict__ Cim,
                       const float* __restrict__ Dp,
                       const float* __restrict__ ln2w, const float* __restrict__ ln2b,
                       const float* __restrict__ Wenc, const float* __restrict__ Wdec,
                       float* __restrict__ xout)
{
    int tok = blockIdx.x, t = threadIdx.x;
    __shared__ float sre[PDIM], sim[PDIM];
    __shared__ float sx1[NDIM], sfx2[NDIM], sval[NDIM], sh[2 * NDIM];
    __shared__ float red[2];
    sre[t] = xsre[(size_t)tok * PDIM + t];
    sim[t] = xsim[(size_t)tok * PDIM + t];
    __syncthreads();
    if (t < NDIM) {
        const float* cr = Cre + t * PDIM;
        const float* ci = Cim + t * PDIM;
        float acc = 0.f;
        #pragma unroll 8
        for (int p = 0; p < PDIM; ++p) acc += cr[p] * sre[p] - ci[p] * sim[p];
        float u = fx[(size_t)tok * NDIM + t];
        float y = 2.0f * acc + Dp[t] * u;
        sx1[t] = gelu_exact(y) + u;
    }
    __syncthreads();
    if (t == 0) {
        float m = 0.f;
        for (int h = 0; h < NDIM; ++h) m += sx1[h];
        m *= (1.0f / NDIM);
        float var = 0.f;
        for (int h = 0; h < NDIM; ++h) { float d = sx1[h] - m; var += d * d; }
        var *= (1.0f / NDIM);
        red[0] = m; red[1] = rsqrtf(var + 1e-5f);
    }
    __syncthreads();
    if (t < NDIM) sfx2[t] = (sx1[t] - red[0]) * red[1] * ln2w[t] + ln2b[t];
    __syncthreads();
    {   // h = fx2 @ Wenc   (64 -> 128)
        float acc = 0.f;
        #pragma unroll 8
        for (int h = 0; h < NDIM; ++h) acc += sfx2[h] * Wenc[h * (2 * NDIM) + t];
        sh[t] = acc;
    }
    __syncthreads();
    if (t < NDIM) sval[t] = sh[t] * gelu_exact(sh[NDIM + t]);   // v * gelu(g)
    __syncthreads();
    if (t < NDIM) {
        float acc = 0.f;
        #pragma unroll 8
        for (int h = 0; h < NDIM; ++h) acc += sval[h] * Wdec[h * NDIM + t];
        xout[(size_t)tok * NDIM + t] = acc + sfx2[t];
    }
}

// ---------------------------------------------------------------------------
// h1 = softplus(x @ W1 + b1), stored bf16. One block (128 thr) per token.
// ---------------------------------------------------------------------------
__global__ void k_head1(const float* __restrict__ x,
                        const float* __restrict__ W1, const float* __restrict__ b1,
                        unsigned short* __restrict__ h1)
{
    int tok = blockIdx.x, t = threadIdx.x;
    __shared__ float sx[NDIM];
    if (t < NDIM) sx[t] = x[(size_t)tok * NDIM + t];
    __syncthreads();
    float acc = b1[t];
    #pragma unroll 8
    for (int h = 0; h < NDIM; ++h) acc += sx[h] * W1[h * HDIM + t];
    float sp = (acc > 30.f) ? acc : log1pf(expf(acc));
    h1[(size_t)tok * HDIM + t] = f2bf(sp);
}

// ---------------------------------------------------------------------------
// Transpose + convert toA_w2 [H=128 x 4096] fp32 -> W2T [4096 x 128] bf16.
// ---------------------------------------------------------------------------
__global__ void k_convW2(const float* __restrict__ W2, unsigned short* __restrict__ W2T)
{
    int n = blockIdx.x, k = threadIdx.x;                 // n<4096, k<128
    W2T[(size_t)n * HDIM + k] = f2bf(W2[(size_t)k * NN + n]);
}

// ---------------------------------------------------------------------------
// WMMA GEMM: A_raw[16384 x 4096] = h1[16384 x 128] @ W2T^T, + b2, * mask.
// Block = 256 thr (8 waves); wave w owns one 16x16 tile; K=128 -> 4 WMMAs.
// Lane layout per CDNA5 16-bit A-fragment spec: lanes 0-15 K{0-7,16-23},
// lanes 16-31 K{8-15,24-31}; B mirrors A on the transposed weight rows.
// ---------------------------------------------------------------------------
__global__ void k_gemm_head2(const unsigned short* __restrict__ h1,
                             const unsigned short* __restrict__ w2t,
                             const float* __restrict__ b2,
                             const float* __restrict__ mask,
                             float* __restrict__ outA)
{
    int lane = threadIdx.x & 31;
    int wave = threadIdx.x >> 5;
    int rowBase = blockIdx.x * 16;                        // token-row tile
    int colBase = (blockIdx.y * 8 + wave) * 16;           // output-col tile
    int halfSel = lane >> 4;                              // 0 | 1
    int mr      = lane & 15;
    int kOff    = halfSel * 8;

    const unsigned short* aRow = h1  + (size_t)(rowBase + mr) * HDIM;
    const unsigned short* bRow = w2t + (size_t)(colBase + mr) * HDIM;

    v8f acc = {};
    #pragma unroll
    for (int k0 = 0; k0 < HDIM; k0 += 32) {
        BfFrag a, b;
        a.q[0] = *(const uint4*)(aRow + k0 + kOff);
        a.q[1] = *(const uint4*)(aRow + k0 + kOff + 16);
        b.q[0] = *(const uint4*)(bRow + k0 + kOff);
        b.q[1] = *(const uint4*)(bRow + k0 + kOff + 16);
        acc = __builtin_amdgcn_wmma_f32_16x16x32_bf16(
                  false, a.v, false, b.v, (short)0, acc, false, false);
    }
    #pragma unroll
    for (int r = 0; r < 8; ++r) {
        int m = halfSel * 8 + r;
        int n = lane & 15;
        int c = colBase + n;
        size_t idx = (size_t)(rowBase + m) * NN + c;
        outA[idx] = (acc[r] + b2[c]) * mask[c];
    }
}

// ---------------------------------------------------------------------------
// Per-token Gauss-Jordan: A_bar = (I - 0.5 dt A)^-1 (I + 0.5 dt A), in place.
// One block (64 thr) per token. A tile staged into LDS via TDM when available.
// M,R in LDS (stride 65 = conflict-free). I - 0.5 dt A is strongly
// diag-dominant (|0.5 dt A| << 1): no pivoting needed.
// ---------------------------------------------------------------------------
__global__ void k_solve(float* __restrict__ bufA, const float* __restrict__ dt)
{
    int tok = blockIdx.x, t = threadIdx.x;                // t < 64
    __shared__ float sA[NN];
    __shared__ float M[NDIM][NDIM + 1];
    __shared__ float R[NDIM][NDIM + 1];
    __shared__ float sinv;
    float* Atok = bufA + (size_t)tok * NN;
#if USE_TDM
    if (t < 32) {                                         // wave 0 issues DMA
        tdm_load_f32(Atok, sA, NN);
        __builtin_amdgcn_s_wait_tensorcnt(0);
    }
    __syncthreads();
#else
    for (int i = 0; i < NDIM; ++i) sA[i * NDIM + t] = Atok[i * NDIM + t];
    __syncthreads();
#endif
    float hdt = 0.5f * dt[tok];
    for (int i = 0; i < NDIM; ++i) {
        float a  = sA[i * NDIM + t] * hdt;
        float id = (i == t) ? 1.0f : 0.0f;
        M[i][t] = id - a;
        R[i][t] = id + a;
    }
    __syncthreads();
    for (int k = 0; k < NDIM; ++k) {
        if (t == k) sinv = 1.0f / M[k][k];
        __syncthreads();
        M[k][t] *= sinv;
        R[k][t] *= sinv;
        __syncthreads();
        if (t != k) {
            float f = M[t][k];
            #pragma unroll 8
            for (int j = 0; j < NDIM; ++j) {
                M[t][j] -= f * M[k][j];
                R[t][j] -= f * R[k][j];
            }
        }
        __syncthreads();
    }
    for (int i = 0; i < NDIM; ++i)
        Atok[i * NDIM + t] = R[i][t];
}

// ---------------------------------------------------------------------------
// Sequential trajectory: z_l = A_bar[b,l-1] @ z_{l-1}. One block per batch.
// TDM double-buffers the next step's 16KB matrix into LDS while the current
// matvec runs; LDS reads skewed ((j+t)&63) for bank-conflict-free access.
// ---------------------------------------------------------------------------
__global__ void k_zscan(const float* __restrict__ bufA,
                        const float* __restrict__ zin,
                        float* __restrict__ out)
{
    int b = blockIdx.x, t = threadIdx.x;                  // t < 64
    __shared__ float z[NDIM], zn[NDIM];
    size_t ob = (size_t)b * LSEQ * NDIM;
    z[t] = zin[ob + t];
    out[ob + t] = z[t];
#if USE_TDM
    __shared__ float mat[2][NN];
    if (t < 32) {                                         // preload m=0
        tdm_load_f32(bufA + ((size_t)b * LSEQ) * NN, mat[0], NN);
        __builtin_amdgcn_s_wait_tensorcnt(0);
    }
    __syncthreads();
    for (int l = 1; l < LSEQ; ++l) {
        int cur = (l - 1) & 1;
        if (t < 32 && l < LSEQ - 1)                       // prefetch m=l
            tdm_load_f32(bufA + ((size_t)b * LSEQ + l) * NN, mat[l & 1], NN);
        float acc = 0.f;
        #pragma unroll 8
        for (int j = 0; j < NDIM; ++j) {
            int jj = (j + t) & (NDIM - 1);
            acc += mat[cur][t * NDIM + jj] * z[jj];
        }
        zn[t] = acc;
        __syncthreads();
        z[t] = zn[t];
        out[ob + (size_t)l * NDIM + t] = z[t];
        if (t < 32) __builtin_amdgcn_s_wait_tensorcnt(0); // prefetch landed
        __syncthreads();
    }
#else
    __syncthreads();
    for (int l = 1; l < LSEQ; ++l) {
        const float* Ab = bufA + ((size_t)b * LSEQ + (l - 1)) * NN;
        __builtin_prefetch(Ab + NN + t * NDIM, 0, 0);     // next step's row
        float acc = 0.f;
        #pragma unroll 8
        for (int j = 0; j < NDIM; ++j) acc += Ab[t * NDIM + j] * z[j];
        zn[t] = acc;
        __syncthreads();
        z[t] = zn[t];
        out[ob + (size_t)l * NDIM + t] = z[t];
        __syncthreads();
    }
#endif
}

// ---------------------------------------------------------------------------
extern "C" void kernel_launch(void* const* d_in, const int* in_sizes, int n_in,
                              void* d_out, int out_size, void* d_ws, size_t ws_size,
                              hipStream_t stream)
{
    const float* z_input  = (const float*)d_in[0];
    const float* dt       = (const float*)d_in[1];
    const float* ln1_w    = (const float*)d_in[2];
    const float* ln1_b    = (const float*)d_in[3];
    const float* Lam_re   = (const float*)d_in[4];
    const float* Lam_im   = (const float*)d_in[5];
    const float* B_re     = (const float*)d_in[6];
    const float* B_im     = (const float*)d_in[7];
    const float* C_re     = (const float*)d_in[8];
    const float* C_im     = (const float*)d_in[9];
    const float* Dd       = (const float*)d_in[10];
    const float* log_step = (const float*)d_in[11];
    const float* ln2_w    = (const float*)d_in[12];
    const float* ln2_b    = (const float*)d_in[13];
    const float* ff_enc_w = (const float*)d_in[14];
    const float* ff_dec_w = (const float*)d_in[15];
    const float* toA_w1   = (const float*)d_in[16];
    const float* toA_b1   = (const float*)d_in[17];
    const float* toA_w2   = (const float*)d_in[18];
    const float* toA_b2   = (const float*)d_in[19];
    const float* mask_A   = (const float*)d_in[20];
    float* out = (float*)d_out;

    // ---- workspace carve-up (256B aligned) ----
    char* w = (char*)d_ws;
    size_t off = 0;
    auto carve = [&](size_t bytes) { char* p = w + off; off += (bytes + 255) & ~(size_t)255; return p; };
    float*          bufA   = (float*)         carve((size_t)NTOK * NN * sizeof(float));   // 268 MB
    float*          xA     = (float*)         carve((size_t)NTOK * NDIM * sizeof(float));
    float*          xB     = (float*)         carve((size_t)NTOK * NDIM * sizeof(float));
    float*          fxbuf  = (float*)         carve((size_t)NTOK * NDIM * sizeof(float));
    float*          bur    = (float*)         carve((size_t)NTOK * PDIM * sizeof(float));
    float*          bui    = (float*)         carve((size_t)NTOK * PDIM * sizeof(float));
    float*          Lbr    = (float*)         carve(PDIM * sizeof(float));
    float*          Lbi    = (float*)         carve(PDIM * sizeof(float));
    float*          Bbr    = (float*)         carve((size_t)PDIM * NDIM * sizeof(float));
    float*          Bbi    = (float*)         carve((size_t)PDIM * NDIM * sizeof(float));
    unsigned short* h1buf  = (unsigned short*)carve((size_t)NTOK * HDIM * sizeof(unsigned short));
    unsigned short* W2T    = (unsigned short*)carve((size_t)NN * HDIM * sizeof(unsigned short));
    (void)ws_size; (void)in_sizes; (void)n_in; (void)out_size;

    // ---- two S5 blocks ----
    const float* xcur = z_input;
    float* xbufs[2] = { xA, xB };
    for (int i = 0; i < 2; ++i) {
        k_prep<<<1, PDIM, 0, stream>>>(Lam_re + i * PDIM, Lam_im + i * PDIM,
                                       log_step + i * PDIM,
                                       B_re + (size_t)i * PDIM * NDIM,
                                       B_im + (size_t)i * PDIM * NDIM,
                                       Lbr, Lbi, Bbr, Bbi);
        k_layernorm<<<NTOK, NDIM, 0, stream>>>(xcur, ln1_w + i * NDIM, ln1_b + i * NDIM, fxbuf);
        k_bu<<<NTOK, PDIM, 0, stream>>>(fxbuf, Bbr, Bbi, bur, bui);
        k_s5scan<<<BSZ, PDIM, 0, stream>>>(Lbr, Lbi, bur, bui);
        k_post<<<NTOK, PDIM, 0, stream>>>(bur, bui, fxbuf,
                                          C_re + (size_t)i * NDIM * PDIM,
                                          C_im + (size_t)i * NDIM * PDIM,
                                          Dd + i * NDIM,
                                          ln2_w + i * NDIM, ln2_b + i * NDIM,
                                          ff_enc_w + (size_t)i * NDIM * 2 * NDIM,
                                          ff_dec_w + (size_t)i * NDIM * NDIM,
                                          xbufs[i]);
        xcur = xbufs[i];
    }

    // ---- head: softplus MLP -> A, via bf16 WMMA GEMM ----
    k_head1<<<NTOK, HDIM, 0, stream>>>(xcur, toA_w1, toA_b1, h1buf);
    k_convW2<<<NN, HDIM, 0, stream>>>(toA_w2, W2T);
    dim3 ggrid(NTOK / 16, NN / (16 * 8));                 // 1024 x 32
    k_gemm_head2<<<ggrid, 256, 0, stream>>>(h1buf, W2T, toA_b2, mask_A, bufA);

    // ---- implicit midpoint solve + sequential trajectory ----
    k_solve<<<NTOK, NDIM, 0, stream>>>(bufA, dt);
    k_zscan<<<BSZ, NDIM, 0, stream>>>(bufA, z_input, out);
}
